// CriticGraphPolicy_89077621719076
// MI455X (gfx1250) — matrix operand order repaired and checked by reading, not compiled, not run on hardware
//
#include <hip/hip_runtime.h>

// ---------------------------------------------------------------------------
// CriticGraphPolicy for MI455X (gfx1250), wave32 + v_wmma_f32_16x16x32_f16.
// Each wave owns 16 batch rows through the whole 20-limb sequence; messages
// are wave-private LDS (6 live slots). All GEMMs run on f16 WMMA with f32
// accumulate; odd K columns (action / x scalars) are rank-1 epilogue updates.
// Round-4 changes: explicit depth-1 software pipeline in gemm4 (A + 4 B
// fragments for k-step ks+1 are loaded before the WMMAs of ks issue, so each
// dscnt wait is covered by a full 4-WMMA group); dropped 20 hoisted constant
// VGPRs (cQB2/cQW3/cMB3 reloaded in epilogues) to give the RA slack for the
// double-buffered fragments.
// ---------------------------------------------------------------------------

typedef _Float16 h8  __attribute__((ext_vector_type(8)));
typedef _Float16 h16 __attribute__((ext_vector_type(16)));
typedef float    f8  __attribute__((ext_vector_type(8)));

#define BATCH      32768
#define NUM_LIMBS  20

// live-range-packed message slot per internal node 0..9
__constant__ int c_SLOT[10] = {0, 1, 2, 0, 3, 1, 4, 2, 5, 0};

// ---- LDS layout (bytes) ----------------------------------------------------
#define OFF_WQ1T  0        // [128][72]  f16  (q_w1 rows 0..31,33..64, transposed)
#define OFF_WQ2T  18432    // [128][136] f16
#define OFF_WM1T  53248    // [128][40]  f16  (m_w1 rows 1..32, transposed)
#define OFF_WM2T  63488    // [128][136] f16
#define OFF_WM3T  98304    // [64][136]  f16
#define OFF_QW1A  115712   // [128] f32  (q_w1 row 32 = action weight)
#define OFF_QW3   116224   // [128] f32
#define OFF_MW1X  116736   // [128] f32  (m_w1 row 0 = x weight)
#define OFF_QB1   117248   // [128] f32
#define OFF_QB2   117760   // [128] f32
#define OFF_MB1   118272   // [128] f32
#define OFF_MB2   118784   // [128] f32
#define OFF_MB3   119296   // [64]  f32
#define OFF_HB0   119552   // 8 waves x [16][136] f16 activation ping buffer
#define OFF_MSG   154368   // 8 waves x 6 slots x [16][72] f16 messages
#define OFF_HB1   264960   // 8 waves x [16][136] f16 activation pong buffer
#define SMEM_BYTES 299776

// cross-lane xor-add within 16-lane groups (wave32, ds_swizzle group-of-32)
#define SWZ_XOR(v, m) \
  __int_as_float(__builtin_amdgcn_ds_swizzle(__float_as_int(v), (((m) << 10) | 0x1f)))

__device__ __forceinline__ float red16(float v) {
  v += SWZ_XOR(v, 1);
  v += SWZ_XOR(v, 2);
  v += SWZ_XOR(v, 4);
  v += SWZ_XOR(v, 8);
  return v;
}

#define TANH_K 2.8853900817779268f   // 2*log2(e)

// A fragment (16x32 f16, MxK) from row-major LDS tile.
// lane<16: elems 0..7 = K k0+0..7, elems 8..15 = K k0+16..23 (row = lane)
// lane>=16: shifted by 8 in K (per ISA 16-bit A layout table).
__device__ __forceinline__ h16 fragA(const _Float16* buf, int pitch, int k0, int lane) {
  const int row = lane & 15, hf = lane >> 4;
  const _Float16* p = buf + row * pitch + k0 + hf * 8;
  h8 lo = *(const h8*)p;
  h8 hi = *(const h8*)(p + 16);
  return __builtin_shufflevector(lo, hi, 0,1,2,3,4,5,6,7,8,9,10,11,12,13,14,15);
}

// B fragment (32x16 f16, KxN) from N-major (transposed) weight tile:
// lane<16: elems 0..7 = K k0+0..7, elems 8..15 = K k0+16..23 (column N=lane);
// lane>=16: shifted by 8 in K (same K order as fragA).
__device__ __forceinline__ h16 fragB(const _Float16* buf, int pitch, int n0, int k0, int lane) {
  const int row = n0 + (lane & 15), hf = lane >> 4;
  const _Float16* p = buf + row * pitch + k0 + hf * 8;
  h8 lo = *(const h8*)p;
  h8 hi = *(const h8*)(p + 16);
  return __builtin_shufflevector(lo, hi, 0,1,2,3,4,5,6,7,8,9,10,11,12,13,14,15);
}

__device__ __forceinline__ f8 wmma16(h16 a, h16 b, f8 c) {
  return __builtin_amdgcn_wmma_f32_16x16x32_f16(false, a, false, b, (short)0, c, false, false);
}

// One 4-N-tile GEMM pass with explicit depth-1 software pipelining:
// fragments for k-step ks+1 are loaded before the WMMAs of ks issue, so the
// dscnt wait for them is covered by a full 4-WMMA XDL group.
template<int KS>
__device__ __forceinline__ void gemm4(const _Float16* abuf,
                                      const _Float16* wbuf, int wpitch,
                                      int ntile0, int lane, f8* acc)
{
  h16 a  = fragA(abuf, 136, 0, lane);
  h16 b0 = fragB(wbuf, wpitch, (ntile0 + 0) * 16, 0, lane);
  h16 b1 = fragB(wbuf, wpitch, (ntile0 + 1) * 16, 0, lane);
  h16 b2 = fragB(wbuf, wpitch, (ntile0 + 2) * 16, 0, lane);
  h16 b3 = fragB(wbuf, wpitch, (ntile0 + 3) * 16, 0, lane);
#pragma unroll
  for (int ks = 0; ks < KS; ++ks) {
    const bool more = (ks + 1) < KS;
    const int kn = more ? (ks + 1) * 32 : 0;
    h16 an = more ? fragA(abuf, 136, kn, lane) : a;
    h16 c0 = more ? fragB(wbuf, wpitch, (ntile0 + 0) * 16, kn, lane) : b0;
    h16 c1 = more ? fragB(wbuf, wpitch, (ntile0 + 1) * 16, kn, lane) : b1;
    h16 c2 = more ? fragB(wbuf, wpitch, (ntile0 + 2) * 16, kn, lane) : b2;
    h16 c3 = more ? fragB(wbuf, wpitch, (ntile0 + 3) * 16, kn, lane) : b3;
    acc[0] = wmma16(a, b0, acc[0]);
    acc[1] = wmma16(a, b1, acc[1]);
    acc[2] = wmma16(a, b2, acc[2]);
    acc[3] = wmma16(a, b3, acc[3]);
    a = an; b0 = c0; b1 = c1; b2 = c2; b3 = c3;
  }
}

__global__ __launch_bounds__(256) void critic_graph_kernel(
    const float* __restrict__ state, const float* __restrict__ action,
    const float* __restrict__ qw1, const float* __restrict__ qb1,
    const float* __restrict__ qw2, const float* __restrict__ qb2,
    const float* __restrict__ qw3, const float* __restrict__ qb3,
    const float* __restrict__ mw1, const float* __restrict__ mb1,
    const float* __restrict__ mw2, const float* __restrict__ mb2,
    const float* __restrict__ mw3, const float* __restrict__ mb3,
    float* __restrict__ out)
{
  extern __shared__ __attribute__((aligned(16))) char smem[];
  _Float16* WQ1T = (_Float16*)(smem + OFF_WQ1T);
  _Float16* WQ2T = (_Float16*)(smem + OFF_WQ2T);
  _Float16* WM1T = (_Float16*)(smem + OFF_WM1T);
  _Float16* WM2T = (_Float16*)(smem + OFF_WM2T);
  _Float16* WM3T = (_Float16*)(smem + OFF_WM3T);
  float* sQW1A = (float*)(smem + OFF_QW1A);
  float* sQW3  = (float*)(smem + OFF_QW3);
  float* sMW1X = (float*)(smem + OFF_MW1X);
  float* sQB1  = (float*)(smem + OFF_QB1);
  float* sQB2  = (float*)(smem + OFF_QB2);
  float* sMB1  = (float*)(smem + OFF_MB1);
  float* sMB2  = (float*)(smem + OFF_MB2);
  float* sMB3  = (float*)(smem + OFF_MB3);
  _Float16* HB0 = (_Float16*)(smem + OFF_HB0);
  _Float16* HB1 = (_Float16*)(smem + OFF_HB1);
  _Float16* MSG = (_Float16*)(smem + OFF_MSG);

  const int tid = threadIdx.x;

  // ---- one-time weight staging: transpose to N-major, convert to f16 ------
  for (int idx = tid; idx < 128 * 64; idx += 256) {
    int n = idx >> 6, k = idx & 63;                 // skip q_w1 row 32 (action)
    WQ1T[n * 72 + k] = (_Float16)qw1[(k < 32 ? k : k + 1) * 128 + n];
  }
  for (int idx = tid; idx < 128 * 128; idx += 256) {
    int n = idx >> 7, k = idx & 127;
    WQ2T[n * 136 + k] = (_Float16)qw2[k * 128 + n];
    WM2T[n * 136 + k] = (_Float16)mw2[k * 128 + n];
  }
  for (int idx = tid; idx < 128 * 32; idx += 256) {
    int n = idx >> 5, k = idx & 31;                 // m_w1 rows 1..32
    WM1T[n * 40 + k] = (_Float16)mw1[(k + 1) * 128 + n];
  }
  for (int idx = tid; idx < 64 * 128; idx += 256) {
    int n = idx >> 7, k = idx & 127;
    WM3T[n * 136 + k] = (_Float16)mw3[k * 64 + n];
  }
  for (int idx = tid; idx < 128; idx += 256) {
    sQW1A[idx] = qw1[32 * 128 + idx];
    sQW3[idx]  = qw3[idx];
    sMW1X[idx] = mw1[idx];
    sQB1[idx]  = qb1[idx];
    sQB2[idx]  = qb2[idx];
    sMB1[idx]  = mb1[idx];
    sMB2[idx]  = mb2[idx];
    if (idx < 64) sMB3[idx] = mb3[idx];
  }
  __syncthreads();   // only barrier needed: main loop is wave-private

  const int lane = tid & 31;
  const int wave = tid >> 5;
  const int hf = lane >> 4;      // C-fragment half: M offset 0 or 8
  const int rs = lane & 15;      // C-fragment N within tile
  const int r0 = blockIdx.x * 128 + wave * 16;      // wave's 16 batch rows
  _Float16* hb0  = HB0 + wave * (16 * 136);         // ping
  _Float16* hb1  = HB1 + wave * (16 * 136);         // pong
  _Float16* msgw = MSG + wave * (6 * 16 * 72);
  const float qb3v = qb3[0];

  // ---- hoist the hottest limb-invariant epilogue constants ----------------
  float cQB1[8], cQW1A[8], cMB1[8], cMW1X[8], cMB2[8];
#pragma unroll
  for (int nt = 0; nt < 8; ++nt) {
    const int n = nt * 16 + rs;
    cQB1[nt]  = sQB1[n];
    cQW1A[nt] = sQW1A[n];
    cMB1[nt]  = sMB1[n];
    cMW1X[nt] = sMW1X[n];
    cMB2[nt]  = sMB2[n];
  }

  float xs_sum[8];
#pragma unroll
  for (int r = 0; r < 8; ++r) xs_sum[r] = 0.f;

  for (int i = 0; i < NUM_LIMBS; ++i) {
    const int p   = (i - 1) >> 1;                 // parent (valid for i>=1)
    const int cix = (i >= 1) ? ((i & 1) ^ 1) : 0; // child index: odd->0, even->1

    // ---- stage A = [state_i (32) | msg_in (32)] into hb0, cols 0..63 ------
    {
      const int m = rs, cb = hf * 16;
      const float4* gs = (const float4*)(state + (size_t)(r0 + m) * 640 + i * 32 + cb);
      float4 s0 = gs[0], s1 = gs[1], s2 = gs[2], s3 = gs[3];
      h8 v0 = { (_Float16)s0.x, (_Float16)s0.y, (_Float16)s0.z, (_Float16)s0.w,
                (_Float16)s1.x, (_Float16)s1.y, (_Float16)s1.z, (_Float16)s1.w };
      h8 v1 = { (_Float16)s2.x, (_Float16)s2.y, (_Float16)s2.z, (_Float16)s2.w,
                (_Float16)s3.x, (_Float16)s3.y, (_Float16)s3.z, (_Float16)s3.w };
      *(h8*)(hb0 + m * 136 + cb)     = v0;
      *(h8*)(hb0 + m * 136 + cb + 8) = v1;
      if (i > 0) {
        const _Float16* mp = msgw + c_SLOT[p] * (16 * 72) + m * 72 + cix * 32 + cb;
        *(h8*)(hb0 + m * 136 + 32 + cb)     = *(const h8*)mp;
        *(h8*)(hb0 + m * 136 + 32 + cb + 8) = *(const h8*)(mp + 8);
      } else {
        h8 z = {};
        *(h8*)(hb0 + m * 136 + 32 + cb)     = z;
        *(h8*)(hb0 + m * 136 + 32 + cb + 8) = z;
      }
      if (i + 1 < NUM_LIMBS)
        __builtin_prefetch(state + (size_t)(r0 + m) * 640 + (i + 1) * 32 + cb, 0, 0);
    }

    // action scalars for this wave's 16 rows (m = r + 8*hf)
    float actv[8];
#pragma unroll
    for (int r = 0; r < 8; ++r)
      actv[r] = action[(size_t)i * BATCH + r0 + r + hf * 8];

    // ---- Q layer1: hb0 -> hb1, + act (rank-1) + bias, relu ----------------
#pragma unroll
    for (int ph = 0; ph < 2; ++ph) {
      f8 acc[4] = {};
      gemm4<2>(hb0, WQ1T, 72, ph * 4, lane, acc);
#pragma unroll
      for (int j = 0; j < 4; ++j) {
        const int nt = ph * 4 + j;
        const int n = nt * 16 + rs;
#pragma unroll
        for (int r = 0; r < 8; ++r) {
          float v = fmaxf(acc[j][r] + cQB1[nt] + actv[r] * cQW1A[nt], 0.f);
          hb1[(r + hf * 8) * 136 + n] = (_Float16)v;
        }
      }
    }

    // ---- Q layer2 (hb1, WMMA) + layer3 (N=1) fused as VALU dot ------------
    float part[8] = {0.f, 0.f, 0.f, 0.f, 0.f, 0.f, 0.f, 0.f};
#pragma unroll
    for (int ph = 0; ph < 2; ++ph) {
      f8 acc[4] = {};
      gemm4<4>(hb1, WQ2T, 136, ph * 4, lane, acc);
#pragma unroll
      for (int j = 0; j < 4; ++j) {
        const int n = (ph * 4 + j) * 16 + rs;
        const float b2 = sQB2[n], w3 = sQW3[n];
#pragma unroll
        for (int r = 0; r < 8; ++r)
          part[r] += fmaxf(acc[j][r] + b2, 0.f) * w3;
      }
    }
    float xv[8];
#pragma unroll
    for (int r = 0; r < 8; ++r) {
      xv[r] = red16(part[r]) + qb3v;   // replicated across the 16-lane group
      xs_sum[r] += xv[r];
    }

    // ---- M-net only for internal nodes (limbs 0..9) -----------------------
    if (i < 10) {
      // A = tanh(msg_in) fragment (zero for root). Phased tanh for ILP.
      h16 ta = {};
      if (i > 0) {
        const _Float16* mp = msgw + c_SLOT[p] * (16 * 72) + rs * 72 + cix * 32 + hf * 8;
        h8 lo = *(const h8*)mp;
        h8 hi = *(const h8*)(mp + 16);
        float ex[16];
#pragma unroll
        for (int j = 0; j < 8; ++j) {
          float u0 = fminf(fmaxf((float)lo[j], -15.f), 15.f);
          float u1 = fminf(fmaxf((float)hi[j], -15.f), 15.f);
          ex[j]     = __builtin_amdgcn_exp2f(u0 * TANH_K);
          ex[j + 8] = __builtin_amdgcn_exp2f(u1 * TANH_K);
        }
#pragma unroll
        for (int j = 0; j < 16; ++j)
          ta[j] = (_Float16)((ex[j] - 1.f) * __builtin_amdgcn_rcpf(ex[j] + 1.f));
      }
      // tanh(x) rank-1 term, phased
      float txe[8], tx[8];
#pragma unroll
      for (int r = 0; r < 8; ++r) {
        float u = fminf(fmaxf(xv[r], -15.f), 15.f);
        txe[r] = __builtin_amdgcn_exp2f(u * TANH_K);
      }
#pragma unroll
      for (int r = 0; r < 8; ++r)
        tx[r] = (txe[r] - 1.f) * __builtin_amdgcn_rcpf(txe[r] + 1.f);

      // ---- M layer1: (tanh frag) -> hb0 -------------------------------------
#pragma unroll
      for (int ph = 0; ph < 2; ++ph) {
        f8 acc[4] = {};
        if (i > 0) {
          h16 b0 = fragB(WM1T, 40, (ph * 4 + 0) * 16, 0, lane);
          h16 b1 = fragB(WM1T, 40, (ph * 4 + 1) * 16, 0, lane);
          h16 b2 = fragB(WM1T, 40, (ph * 4 + 2) * 16, 0, lane);
          h16 b3 = fragB(WM1T, 40, (ph * 4 + 3) * 16, 0, lane);
          acc[0] = wmma16(ta, b0, acc[0]);
          acc[1] = wmma16(ta, b1, acc[1]);
          acc[2] = wmma16(ta, b2, acc[2]);
          acc[3] = wmma16(ta, b3, acc[3]);
        }
#pragma unroll
        for (int j = 0; j < 4; ++j) {
          const int nt = ph * 4 + j;
          const int n = nt * 16 + rs;
#pragma unroll
          for (int r = 0; r < 8; ++r) {
            float v = fmaxf(acc[j][r] + cMB1[nt] + tx[r] * cMW1X[nt], 0.f);
            hb0[(r + hf * 8) * 136 + n] = (_Float16)v;
          }
        }
      }

      // ---- M layer2: hb0 -> hb1 --------------------------------------------
#pragma unroll
      for (int ph = 0; ph < 2; ++ph) {
        f8 acc[4] = {};
        gemm4<4>(hb0, WM2T, 136, ph * 4, lane, acc);
#pragma unroll
        for (int j = 0; j < 4; ++j) {
          const int nt = ph * 4 + j;
          const int n = nt * 16 + rs;
#pragma unroll
          for (int r = 0; r < 8; ++r)
            hb1[(r + hf * 8) * 136 + n] = (_Float16)fmaxf(acc[j][r] + cMB2[nt], 0.f);
        }
      }

      // ---- M layer3 (N=64, hb1) + L2 normalize -> message slot -------------
      f8 acc4[4] = {};
      gemm4<4>(hb1, WM3T, 136, 0, lane, acc4);
      float ss[8] = {0.f, 0.f, 0.f, 0.f, 0.f, 0.f, 0.f, 0.f};
      float val[4][8];
#pragma unroll
      for (int nt = 0; nt < 4; ++nt) {
        const float b3 = sMB3[nt * 16 + rs];
#pragma unroll
        for (int r = 0; r < 8; ++r) {
          float v = acc4[nt][r] + b3;
          val[nt][r] = v;
          ss[r] += v * v;
        }
      }
      _Float16* md = msgw + c_SLOT[i] * (16 * 72);
#pragma unroll
      for (int r = 0; r < 8; ++r) {
        float s   = red16(ss[r]);
        float nrm = __builtin_amdgcn_sqrtf(s);
        float sc  = __builtin_amdgcn_rcpf(fmaxf(nrm, 1e-12f));
#pragma unroll
        for (int nt = 0; nt < 4; ++nt)
          md[(r + hf * 8) * 72 + nt * 16 + rs] = (_Float16)(val[nt][r] * sc);
      }
    }
  }

  // ---- output: sum of x over limbs, one value per batch row ---------------
  if (rs == 0) {
#pragma unroll
    for (int r = 0; r < 8; ++r)
      out[r0 + r + hf * 8] = xs_sum[r];
  }
}

extern "C" void kernel_launch(void* const* d_in, const int* in_sizes, int n_in,
                              void* d_out, int out_size, void* d_ws, size_t ws_size,
                              hipStream_t stream) {
  (void)in_sizes; (void)n_in; (void)out_size; (void)d_ws; (void)ws_size;
  const float* state = (const float*)d_in[0];
  const float* actn  = (const float*)d_in[1];
  const float* qw1   = (const float*)d_in[2];
  const float* qb1   = (const float*)d_in[3];
  const float* qw2   = (const float*)d_in[4];
  const float* qb2   = (const float*)d_in[5];
  const float* qw3   = (const float*)d_in[6];
  const float* qb3   = (const float*)d_in[7];
  const float* mw1   = (const float*)d_in[8];
  const float* mb1   = (const float*)d_in[9];
  const float* mw2   = (const float*)d_in[10];
  const float* mb2   = (const float*)d_in[11];
  const float* mw3   = (const float*)d_in[12];
  const float* mb3   = (const float*)d_in[13];
  float* out = (float*)d_out;

  (void)hipFuncSetAttribute((const void*)critic_graph_kernel,
                            hipFuncAttributeMaxDynamicSharedMemorySize,
                            SMEM_BYTES);

  dim3 grid(BATCH / 128), block(256);
  critic_graph_kernel<<<grid, block, SMEM_BYTES, stream>>>(
      state, actn, qw1, qb1, qw2, qb2, qw3, qb3,
      mw1, mb1, mw2, mb2, mw3, mb3, out);
}
